// LocalFacialExtractor_39814346834391
// MI455X (gfx1250) — compile-verified, matrix operand order checked by reading, not run on hardware
//
#include <hip/hip_runtime.h>
#include <hip/hip_bf16.h>
#include <math.h>
#include <stdint.h>

// ---------------------------------------------------------------------------
// MI455X (gfx1250) implementation of LocalFacialExtractor.
// All big GEMMs -> bf16 WMMA (v_wmma_f32_16x16x32_bf16), f32 accumulate.
// Weights transposed+converted to bf16 once per launch (Wt[N][K]) so the
// B-operand fragment loads are contiguous. LDS double-buffered staging via
// GLOBAL_LOAD_ASYNC_TO_LDS_B128 (ASYNCcnt) — copies bypass the VGPR file.
// Attention (Lq=37, tiny) is a flash-style VALU kernel.
// ---------------------------------------------------------------------------

typedef __bf16 bf16;
typedef __attribute__((ext_vector_type(16))) __bf16 bf16x16;
typedef __attribute__((ext_vector_type(8)))  __bf16 bf16x8;
typedef __attribute__((ext_vector_type(8)))  float  f32x8;

#define BM 64
#define BN 128
#define BK 32
#define TPB 256

__device__ __forceinline__ float gelu_exact(float v) {
  return 0.5f * v * (1.0f + erff(v * 0.70710678118654752f));
}

// ---- gfx1250 async global->LDS copy (16B per lane), tracked by ASYNCcnt ----
__device__ __forceinline__ void async_copy16(uint32_t lds_addr, const bf16* gp) {
  asm volatile("global_load_async_to_lds_b128 %0, %1, off"
               :
               : "v"(lds_addr), "v"((unsigned long long)(uintptr_t)gp)
               : "memory");
}
__device__ __forceinline__ void async_wait0() {
  asm volatile("s_wait_asynccnt 0x0" ::: "memory");
}

// ---------------- weight transpose+convert: W f32 [S][K][N] -> Wt bf16 [S][N][K]
__global__ void __launch_bounds__(256) k_wt(const float* __restrict__ W,
                                            bf16* __restrict__ Wt, int K, int N) {
  __shared__ float tile[32][33];
  int s = blockIdx.z;
  int n0 = blockIdx.x * 32, k0 = blockIdx.y * 32;
  const float* Ws = W + (size_t)s * K * N;
  bf16* Wts = Wt + (size_t)s * N * K;
  int tx = threadIdx.x & 31, ty = threadIdx.x >> 5;
  for (int i = ty; i < 32; i += 8) {
    int k = k0 + i, n = n0 + tx;
    tile[i][tx] = (k < K && n < N) ? Ws[(size_t)k * N + n] : 0.0f;
  }
  __syncthreads();
  for (int i = ty; i < 32; i += 8) {
    int n = n0 + i, k = k0 + tx;
    if (n < N && k < K) Wts[(size_t)n * K + k] = (bf16)tile[tx][i];
  }
}

// ---------------- plain f32 -> bf16 convert
__global__ void k_cvt(const float* __restrict__ in, bf16* __restrict__ out, long long n) {
  long long i = (long long)blockIdx.x * blockDim.x + threadIdx.x;
  if (i < n) out[i] = (bf16)in[i];
}

// lat[:, :32] -> bf16 [1024][1024]
__global__ void k_gather32(const float* __restrict__ lat, bf16* __restrict__ out) {
  int i = blockIdx.x * blockDim.x + threadIdx.x;
  if (i < 1024 * 1024) {
    int c = i & 1023, r = i >> 10;
    int b = r >> 5, qq = r & 31;
    out[i] = (bf16)lat[((size_t)b * 37 + qq) * 1024 + c];
  }
}

// lat init: rows 0..31 = latents broadcast, 32..36 = xid
__global__ void k_init_lat(const float* __restrict__ latents,
                           const float* __restrict__ xid,
                           float* __restrict__ lat) {
  int i = blockIdx.x * blockDim.x + threadIdx.x;
  const int total = 32 * 37 * 1024;
  if (i < total) {
    int c = i & 1023, r = i >> 10;
    int tok = r % 37, b = r / 37;
    lat[i] = (tok < 32) ? latents[(size_t)tok * 1024 + c]
                        : xid[((size_t)b * 5 + (tok - 32)) * 1024 + c];
  }
}

// ---------------- LayerNorm (+optional leaky-relu), f32 in -> bf16 out
// in row  = in  + b*in_bstride + tok*C
// out row = out + (b*out_brows + out_toff + tok)*C
__global__ void __launch_bounds__(256) k_ln(const float* __restrict__ in,
                                            long long in_bstride,
                                            const float* __restrict__ g,
                                            const float* __restrict__ beta,
                                            int C, int leaky,
                                            bf16* __restrict__ out,
                                            long long out_brows, long long out_toff) {
  int tok = blockIdx.x, b = blockIdx.y;
  const float* row = in + (size_t)b * in_bstride + (size_t)tok * C;
  bf16* orow = out + ((size_t)b * out_brows + (size_t)(out_toff + tok)) * C;
  int tid = threadIdx.x, lane = tid & 31, wv = tid >> 5;
  float s = 0.0f, ss = 0.0f;
  for (int i = tid; i < C; i += 256) { float v = row[i]; s += v; ss += v * v; }
#pragma unroll
  for (int o = 16; o > 0; o >>= 1) { s += __shfl_down(s, o, 32); ss += __shfl_down(ss, o, 32); }
  __shared__ float red[2][8];
  if (lane == 0) { red[0][wv] = s; red[1][wv] = ss; }
  __syncthreads();
  if (tid == 0) {
    float S = 0.0f, SS = 0.0f;
#pragma unroll
    for (int kq = 0; kq < 8; ++kq) { S += red[0][kq]; SS += red[1][kq]; }
    float mu = S / (float)C;
    float var = SS / (float)C - mu * mu;
    red[0][0] = mu;
    red[1][0] = rsqrtf(var + 1e-5f);
  }
  __syncthreads();
  float mu = red[0][0], rstd = red[1][0];
  for (int i = tid; i < C; i += 256) {
    float v = (row[i] - mu) * rstd * g[i] + beta[i];
    if (leaky) v = v > 0.0f ? v : 0.01f * v;
    orow[i] = (bf16)v;
  }
}

// ---------------- bf16 WMMA GEMM ----------------
// Stage the next 64x32 A panel and 128x32 B panel into LDS with async copies.
// Out-of-range rows are CLAMPED (not zeroed): a padded A row only feeds
// C rows >= M and a padded B row only feeds C cols >= N, both of which the
// epilogue masks — so EXEC stays all-1s and no zero-fill path is needed.
__device__ __forceinline__ void stage_async(const bf16* __restrict__ A,
                                            const bf16* __restrict__ Bt,
                                            int M, int N, int K, int bm0, int bn0,
                                            int kk, int tid,
                                            uint32_t ldsA, uint32_t ldsB) {
  int row = tid >> 2, ch = (tid & 3) * 8;   // 16B chunk per lane
  {
    int gr = bm0 + row;
    gr = gr < M ? gr : (M - 1);
    const bf16* p = A + (size_t)gr * K + kk + ch;
    async_copy16(ldsA + (uint32_t)(row * BK + ch) * 2u, p);
    __builtin_prefetch(p + BK, 0, 1);   // global_prefetch_b8, K panel after next
  }
#pragma unroll
  for (int rep = 0; rep < 2; ++rep) {
    int brow = row + rep * 64;
    int gr = bn0 + brow;
    gr = gr < N ? gr : (N - 1);
    const bf16* p = Bt + (size_t)gr * K + kk + ch;
    async_copy16(ldsB + (uint32_t)(brow * BK + ch) * 2u, p);
  }
}

// 16-bit operand fragment: lane (0-15) holds row m=lane, K 0..7 & 16..23;
// lane (16-31) holds K 8..15 & 24..31 (ISA 7.12.2).
__device__ __forceinline__ bf16x16 load_frag(const bf16* __restrict__ rowp, int khalf) {
  bf16x8 lo = *(const bf16x8*)(rowp + khalf * 8);
  bf16x8 hi = *(const bf16x8*)(rowp + 16 + khalf * 8);
  return __builtin_shufflevector(lo, hi, 0, 1, 2, 3, 4, 5, 6, 7,
                                 8, 9, 10, 11, 12, 13, 14, 15);
}

// C[M,N] = A[M,K] @ Bt[N,K]^T (+bias)(+act)(+Res); writes f32 and/or bf16.
__global__ void __launch_bounds__(TPB) k_gemm(const bf16* __restrict__ A,
                                              const bf16* __restrict__ Bt,
                                              const float* __restrict__ bias,
                                              float* __restrict__ Cf,
                                              bf16* __restrict__ Cb,
                                              const float* __restrict__ Res,
                                              int M, int N, int K, int act) {
  __shared__ __align__(16) bf16 As[2][BM][BK];
  __shared__ __align__(16) bf16 Bs[2][BN][BK];
  int tid = threadIdx.x;
  int wave = tid >> 5, lane = tid & 31;
  int wm = wave & 1, wn = wave >> 1;          // 2 x 4 wave grid
  int bm0 = blockIdx.y * BM, bn0 = blockIdx.x * BN;

  // raw LDS byte addresses (flat addr truncates to LDS offset, ISA 10.2)
  uint32_t ldsA0 = (uint32_t)(uintptr_t)(&As[0][0][0]);
  uint32_t ldsB0 = (uint32_t)(uintptr_t)(&Bs[0][0][0]);
  const uint32_t ABUF = (uint32_t)(BM * BK * 2);   // 4 KB
  const uint32_t BBUF = (uint32_t)(BN * BK * 2);   // 8 KB

  const f32x8 zero = {0.f, 0.f, 0.f, 0.f, 0.f, 0.f, 0.f, 0.f};
  f32x8 acc[2][2];
#pragma unroll
  for (int i = 0; i < 2; ++i)
#pragma unroll
    for (int j = 0; j < 2; ++j) acc[i][j] = zero;

  int nk = K / BK;
  stage_async(A, Bt, M, N, K, bm0, bn0, 0, tid, ldsA0, ldsB0);
  async_wait0();
  __syncthreads();

  int mrow = lane & 15, khalf = lane >> 4;
  for (int kb = 0; kb < nk; ++kb) {
    int cur = kb & 1;
    if (kb + 1 < nk) {
      int nxt = cur ^ 1;
      stage_async(A, Bt, M, N, K, bm0, bn0, (kb + 1) * BK, tid,
                  ldsA0 + (uint32_t)nxt * ABUF, ldsB0 + (uint32_t)nxt * BBUF);
    }
    bf16x16 af[2], bfr[2];
#pragma unroll
    for (int sm = 0; sm < 2; ++sm)
      af[sm] = load_frag(&As[cur][wm * 32 + sm * 16 + mrow][0], khalf);
#pragma unroll
    for (int sn = 0; sn < 2; ++sn)
      bfr[sn] = load_frag(&Bs[cur][wn * 32 + sn * 16 + mrow][0], khalf);
#pragma unroll
    for (int sm = 0; sm < 2; ++sm)
#pragma unroll
      for (int sn = 0; sn < 2; ++sn)
        acc[sm][sn] = __builtin_amdgcn_wmma_f32_16x16x32_bf16(
            false, af[sm], false, bfr[sn], (short)0, acc[sm][sn], false, false);
    async_wait0();       // our async fills of the next buffer are done
    __syncthreads();     // everyone's fills are done / reads of cur complete
  }

  // epilogue: C/D layout -> VGPR r: lane<16 M=r, lane>=16 M=8+r; N=lane&15
  int nofs = lane & 15, mhalf = (lane >> 4) * 8;
#pragma unroll
  for (int sm = 0; sm < 2; ++sm) {
#pragma unroll
    for (int sn = 0; sn < 2; ++sn) {
      int n = bn0 + wn * 32 + sn * 16 + nofs;
      if (n >= N) continue;
      float bv = bias ? bias[n] : 0.0f;
#pragma unroll
      for (int r = 0; r < 8; ++r) {
        int m = bm0 + wm * 32 + sm * 16 + mhalf + r;
        if (m >= M) continue;
        float v = acc[sm][sn][r] + bv;
        if (act == 1) v = gelu_exact(v);
        size_t idx = (size_t)m * N + n;
        if (Res) v += Res[idx];
        if (Cf) Cf[idx] = v;
        if (Cb) Cb[idx] = (bf16)v;
      }
    }
  }
}

// ---------------- flash-style attention: grid = B*H, 8 waves, wave per q-row
__global__ void __launch_bounds__(256) k_attn(const bf16* __restrict__ Q,
                                              const bf16* __restrict__ KV,
                                              bf16* __restrict__ O) {
  int bh = blockIdx.x;
  int b = bh >> 4, h = bh & 15;
  const bf16* qbp = Q + (size_t)b * 37 * 1024 + h * 64;
  const bf16* kbp = KV + (size_t)b * 619 * 2048 + h * 64;
  const bf16* vbp = kbp + 1024;
  bf16* obp = O + (size_t)b * 37 * 1024 + h * 64;
  int wave = threadIdx.x >> 5, lane = threadIdx.x & 31;
  for (int qr = wave; qr < 37; qr += 8) {
    const bf16* qp = qbp + (size_t)qr * 1024;
    float q[64];
#pragma unroll
    for (int c = 0; c < 8; ++c) {
      bf16x8 v = *(const bf16x8*)(qp + c * 8);
#pragma unroll
      for (int e = 0; e < 8; ++e) q[c * 8 + e] = (float)v[e];
    }
    float m = -3.0e38f, s = 0.0f, accv[64];
#pragma unroll
    for (int d = 0; d < 64; ++d) accv[d] = 0.0f;
    for (int kr = lane; kr < 619; kr += 32) {
      const bf16* kp = kbp + (size_t)kr * 2048;
      float dot = 0.0f;
#pragma unroll
      for (int c = 0; c < 8; ++c) {
        bf16x8 v = *(const bf16x8*)(kp + c * 8);
#pragma unroll
        for (int e = 0; e < 8; ++e) dot += q[c * 8 + e] * (float)v[e];
      }
      dot *= 0.125f;  // (1/sqrt(sqrt(64)))^2 applied to logits
      float mn = fmaxf(m, dot);
      float fs = __expf(m - mn);
      float p = __expf(dot - mn);
      s = s * fs + p;
      const bf16* vp = vbp + (size_t)kr * 2048;
#pragma unroll
      for (int c = 0; c < 8; ++c) {
        bf16x8 v = *(const bf16x8*)(vp + c * 8);
#pragma unroll
        for (int e = 0; e < 8; ++e)
          accv[c * 8 + e] = accv[c * 8 + e] * fs + p * (float)v[e];
      }
      m = mn;
    }
    float mg = m;
#pragma unroll
    for (int o = 16; o > 0; o >>= 1) mg = fmaxf(mg, __shfl_xor(mg, o, 32));
    float fs = __expf(m - mg);
    float sg = s * fs;
#pragma unroll
    for (int o = 16; o > 0; o >>= 1) sg += __shfl_xor(sg, o, 32);
    float inv = 1.0f / sg;
#pragma unroll
    for (int d = 0; d < 64; ++d) {
      float a = accv[d] * fs;
#pragma unroll
      for (int o = 16; o > 0; o >>= 1) a += __shfl_xor(a, o, 32);
      if (lane == 0) obp[(size_t)qr * 1024 + d] = (bf16)(a * inv);
    }
  }
}

// ---------------------------------------------------------------------------
extern "C" void kernel_launch(void* const* d_in, const int* in_sizes, int n_in,
                              void* d_out, int out_size, void* d_ws, size_t ws_size,
                              hipStream_t stream) {
  (void)in_sizes; (void)n_in; (void)out_size; (void)ws_size;
  const float* x       = (const float*)d_in[0];
  const float* y       = (const float*)d_in[1];
  const float* latents = (const float*)d_in[2];
  const float* projw   = (const float*)d_in[3];
  const float* id_w1 = (const float*)d_in[4];
  const float* id_b1 = (const float*)d_in[5];
  const float* id_g1 = (const float*)d_in[6];
  const float* id_h1 = (const float*)d_in[7];
  const float* id_w2 = (const float*)d_in[8];
  const float* id_b2 = (const float*)d_in[9];
  const float* id_g2 = (const float*)d_in[10];
  const float* id_h2 = (const float*)d_in[11];
  const float* id_w3 = (const float*)d_in[12];
  const float* id_b3 = (const float*)d_in[13];
  const float* map_w1 = (const float*)d_in[14];
  const float* map_b1 = (const float*)d_in[15];
  const float* map_g1 = (const float*)d_in[16];
  const float* map_h1 = (const float*)d_in[17];
  const float* map_w2 = (const float*)d_in[18];
  const float* map_b2 = (const float*)d_in[19];
  const float* map_g2 = (const float*)d_in[20];
  const float* map_h2 = (const float*)d_in[21];
  const float* map_w3 = (const float*)d_in[22];
  const float* map_b3 = (const float*)d_in[23];
  const float* n1g = (const float*)d_in[24];
  const float* n1b = (const float*)d_in[25];
  const float* n2g = (const float*)d_in[26];
  const float* n2b = (const float*)d_in[27];
  const float* wq  = (const float*)d_in[28];
  const float* wkv = (const float*)d_in[29];
  const float* wo  = (const float*)d_in[30];
  const float* ffg = (const float*)d_in[31];
  const float* ffb = (const float*)d_in[32];
  const float* ffw1 = (const float*)d_in[33];
  const float* ffw2 = (const float*)d_in[34];

  char* ws = (char*)d_ws;
  size_t off = 0;
  auto ab = [&](size_t elems, size_t esz) -> char* {
    off = (off + 255) & ~(size_t)255;
    char* p = ws + off;
    off += elems * esz;
    return p;
  };

  // transposed bf16 weights
  bf16* idw1t = (bf16*)ab(1280ull * 1024, 2);
  bf16* idw2t = (bf16*)ab(1024ull * 1024, 2);
  bf16* idw3t = (bf16*)ab(5120ull * 1024, 2);
  bf16* mw1t  = (bf16*)ab(5ull * 1024 * 1024, 2);
  bf16* mw2t  = (bf16*)ab(5ull * 1024 * 1024, 2);
  bf16* mw3t  = (bf16*)ab(5ull * 1024 * 1024, 2);
  bf16* wqT   = (bf16*)ab(10ull * 1024 * 1024, 2);
  bf16* wkvT  = (bf16*)ab(10ull * 2048 * 1024, 2);
  bf16* woT   = (bf16*)ab(10ull * 1024 * 1024, 2);
  bf16* f1T   = (bf16*)ab(10ull * 4096 * 1024, 2);
  bf16* f2T   = (bf16*)ab(10ull * 1024 * 4096, 2);
  bf16* prT   = (bf16*)ab(2048ull * 1024, 2);
  // activations
  bf16*  xb   = (bf16*)ab(32ull * 1280, 2);
  float* hf   = (float*)ab(32ull * 1024, 4);
  bf16*  hb   = (bf16*)ab(32ull * 1024, 2);
  float* xid  = (float*)ab(32ull * 5120, 4);
  float* lat  = (float*)ab(32ull * 37 * 1024, 4);
  bf16*  yb   = (bf16*)ab(18464ull * 1024, 2);
  float* tf   = (float*)ab(18464ull * 1024, 4);
  bf16*  tb   = (bf16*)ab(18464ull * 1024, 2);
  float* vit  = (float*)ab(18464ull * 1024, 4);
  bf16*  latn = (bf16*)ab(1184ull * 1024, 2);
  bf16*  kvin = (bf16*)ab(19808ull * 1024, 2);
  bf16*  qb16 = (bf16*)ab(1184ull * 1024, 2);
  bf16*  kvb  = (bf16*)ab(19808ull * 2048, 2);
  bf16*  ob16 = (bf16*)ab(1184ull * 1024, 2);
  bf16*  ub   = (bf16*)ab(1184ull * 4096, 2);
  bf16*  a32  = (bf16*)ab(1024ull * 1024, 2);

  auto tr = [&](const float* W, bf16* Wt, int K, int N, int S) {
    dim3 g((N + 31) / 32, (K + 31) / 32, S);
    k_wt<<<g, dim3(256), 0, stream>>>(W, Wt, K, N);
  };
  auto gemm = [&](const bf16* A, const bf16* Bt, const float* bias,
                  float* Cf, bf16* Cb, const float* Res,
                  int M, int N, int K, int act) {
    dim3 g((N + BN - 1) / BN, (M + BM - 1) / BM);
    k_gemm<<<g, dim3(TPB), 0, stream>>>(A, Bt, bias, Cf, Cb, Res, M, N, K, act);
  };
  auto ln = [&](const float* in, long long ibs, const float* g_, const float* b_,
                int leaky, bf16* out, long long obrows, long long toff,
                int toks, int batches) {
    dim3 g(toks, batches);
    k_ln<<<g, dim3(256), 0, stream>>>(in, ibs, g_, b_, 1024, leaky, out, obrows, toff);
  };

  // ---- weight prep (once per launch) ----
  tr(id_w1, idw1t, 1280, 1024, 1);
  tr(id_w2, idw2t, 1024, 1024, 1);
  tr(id_w3, idw3t, 1024, 5120, 1);
  tr(map_w1, mw1t, 1024, 1024, 5);
  tr(map_w2, mw2t, 1024, 1024, 5);
  tr(map_w3, mw3t, 1024, 1024, 5);
  tr(wq, wqT, 1024, 1024, 10);
  tr(wkv, wkvT, 1024, 2048, 10);
  tr(wo, woT, 1024, 1024, 10);
  tr(ffw1, f1T, 1024, 4096, 10);
  tr(ffw2, f2T, 4096, 1024, 10);
  tr(projw, prT, 1024, 2048, 1);

  // ---- id mlp3: x -> xid [32,5,1024] ----
  {
    long long n = 32LL * 1280;
    k_cvt<<<(unsigned)((n + 255) / 256), 256, 0, stream>>>(x, xb, n);
  }
  gemm(xb, idw1t, id_b1, hf, nullptr, nullptr, 32, 1024, 1280, 0);
  ln(hf, 0, id_g1, id_h1, 1, hb, 0, 0, 32, 1);
  gemm(hb, idw2t, id_b2, hf, nullptr, nullptr, 32, 1024, 1024, 0);
  ln(hf, 0, id_g2, id_h2, 1, hb, 0, 0, 32, 1);
  gemm(hb, idw3t, id_b3, xid, nullptr, nullptr, 32, 5120, 1024, 0);
  {
    int n = 32 * 37 * 1024;
    k_init_lat<<<(n + 255) / 256, 256, 0, stream>>>(latents, xid, lat);
  }

  for (int i = 0; i < 5; ++i) {
    // vit = mlp3(y[i])
    long long n = 18464LL * 1024;
    k_cvt<<<(unsigned)((n + 255) / 256), 256, 0, stream>>>(y + (size_t)i * n, yb, n);
    gemm(yb, mw1t + (size_t)i * 1024 * 1024, map_b1 + i * 1024, tf, nullptr, nullptr,
         18464, 1024, 1024, 0);
    ln(tf, 0, map_g1 + i * 1024, map_h1 + i * 1024, 1, tb, 0, 0, 18464, 1);
    gemm(tb, mw2t + (size_t)i * 1024 * 1024, map_b2 + i * 1024, tf, nullptr, nullptr,
         18464, 1024, 1024, 0);
    ln(tf, 0, map_g2 + i * 1024, map_h2 + i * 1024, 1, tb, 0, 0, 18464, 1);
    gemm(tb, mw3t + (size_t)i * 1024 * 1024, map_b3 + i * 1024, vit, nullptr, nullptr,
         18464, 1024, 1024, 0);

    for (int j = 2 * i; j < 2 * i + 2; ++j) {
      const float* g1 = n1g + j * 1024; const float* b1 = n1b + j * 1024;
      const float* g2 = n2g + j * 1024; const float* b2 = n2b + j * 1024;
      // kvin = [ln1(xid) ; ln1(vit) ; ln2(lat)] per batch (619 rows)
      ln(xid, 5 * 1024, g1, b1, 0, kvin, 619, 0, 5, 32);
      ln(vit, 577 * 1024, g1, b1, 0, kvin, 619, 5, 577, 32);
      ln(lat, 37 * 1024, g2, b2, 0, kvin, 619, 582, 37, 32);
      ln(lat, 37 * 1024, g2, b2, 0, latn, 37, 0, 37, 32);
      // q / kv projections -> bf16
      gemm(latn, wqT + (size_t)j * 1024 * 1024, nullptr, nullptr, qb16, nullptr,
           1184, 1024, 1024, 0);
      gemm(kvin, wkvT + (size_t)j * 2048 * 1024, nullptr, nullptr, kvb, nullptr,
           19808, 2048, 1024, 0);
      k_attn<<<512, 256, 0, stream>>>(qb16, kvb, ob16);
      // lat = o @ wo + lat
      gemm(ob16, woT + (size_t)j * 1024 * 1024, nullptr, lat, nullptr, lat,
           1184, 1024, 1024, 0);
      // ffwd: lat = gelu(ln(lat) @ w1) @ w2 + lat
      ln(lat, 37 * 1024, ffg + j * 1024, ffb + j * 1024, 0, latn, 37, 0, 37, 32);
      gemm(latn, f1T + (size_t)j * 4096 * 1024, nullptr, nullptr, ub, nullptr,
           1184, 4096, 1024, 1);
      gemm(ub, f2T + (size_t)j * 1024 * 4096, nullptr, lat, nullptr, lat,
           1184, 1024, 4096, 0);
    }
  }

  {
    int n = 1024 * 1024;
    k_gather32<<<(n + 255) / 256, 256, 0, stream>>>(lat, a32);
  }
  gemm(a32, prT, nullptr, (float*)d_out, nullptr, nullptr, 1024, 2048, 1024, 0);
}
